// HungarianMatcher_72155450572906
// MI455X (gfx1250) — matrix operand order repaired and checked by reading, not compiled, not run on hardware
//
#include <hip/hip_runtime.h>
#include <hip/hip_bf16.h>

// ---------------------------------------------------------------------------
// Problem constants (from reference setup_inputs)
// ---------------------------------------------------------------------------
#define BQ    9600     // B*Q predictions (rows)
#define MT    2400     // total targets (cols)
#define CC    92       // classes
#define KPAD  96       // padded K for WMMA (multiple of 32)
#define NITER 5        // n-tiles per wave in the main kernel
#define ROWS_PER_BLOCK (4 * 16 * NITER)   // 4 waveN bands * 16 rows * NITER = 320

typedef __bf16  __attribute__((ext_vector_type(16))) v16bf;
typedef float   __attribute__((ext_vector_type(8)))  v8f;
typedef float   __attribute__((ext_vector_type(4)))  f32x4;

// ---------------------------------------------------------------------------
// Kernel 1: row softmax, store -prob as bf16 into K=96 padded buffer.
// One wave32 per row; 8 rows per 256-thread block.
// ---------------------------------------------------------------------------
__global__ void softmax_negbf16_kernel(const float* __restrict__ logits,
                                       __hip_bfloat16* __restrict__ negprob) {
    const int wv   = threadIdx.x >> 5;
    const int lane = threadIdx.x & 31;
    const int row  = blockIdx.x * 8 + wv;          // grid exactly covers 9600 rows

    const float* src = logits + (size_t)row * CC;
    float a0 = src[lane];
    float a1 = src[lane + 32];
    float a2 = (lane < CC - 64) ? src[lane + 64] : -3.4e38f;

    float mx = fmaxf(fmaxf(a0, a1), a2);
    #pragma unroll
    for (int off = 16; off > 0; off >>= 1) mx = fmaxf(mx, __shfl_xor(mx, off));

    float e0 = __expf(a0 - mx);
    float e1 = __expf(a1 - mx);
    float e2 = (lane < CC - 64) ? __expf(a2 - mx) : 0.0f;

    float s = e0 + e1 + e2;
    #pragma unroll
    for (int off = 16; off > 0; off >>= 1) s += __shfl_xor(s, off);

    const float inv = __builtin_amdgcn_rcpf(s);

    __hip_bfloat16* dst = negprob + (size_t)row * KPAD;
    dst[lane]      = __float2bfloat16(-e0 * inv);
    dst[lane + 32] = __float2bfloat16(-e1 * inv);
    // lanes >= 28 write the K-pad zeros (cols 92..95)
    dst[lane + 64] = __float2bfloat16((lane < CC - 64) ? (-e2 * inv) : 0.0f);
}

// ---------------------------------------------------------------------------
// Kernel 2: precompute xyxy for pred and target boxes (cxcywh input).
// (areas are recomputed on the fly: 1 VALU beats a load slot)
// ---------------------------------------------------------------------------
__global__ void box_prep_kernel(const float* __restrict__ pred_cxcywh,
                                const float* __restrict__ tgt_cxcywh,
                                float* __restrict__ pxyxy,
                                float* __restrict__ txyxy) {
    const int i = blockIdx.x * blockDim.x + threadIdx.x;
    if (i < BQ) {
        f32x4 b = *(const f32x4*)(pred_cxcywh + (size_t)i * 4);
        f32x4 r;
        r.x = b.x - 0.5f * b.z;  r.y = b.y - 0.5f * b.w;
        r.z = b.x + 0.5f * b.z;  r.w = b.y + 0.5f * b.w;
        *(f32x4*)(pxyxy + (size_t)i * 4) = r;
    }
    if (i < MT) {
        f32x4 b = *(const f32x4*)(tgt_cxcywh + (size_t)i * 4);
        f32x4 r;
        r.x = b.x - 0.5f * b.z;  r.y = b.y - 0.5f * b.w;
        r.z = b.x + 0.5f * b.z;  r.w = b.y + 0.5f * b.w;
        *(f32x4*)(txyxy + (size_t)i * 4) = r;
    }
}

// ---------------------------------------------------------------------------
// Kernel 3: fused cost matrix.
//   block = 256 threads = 8 waves; block tile = 320 (N) x 32 (M);
//   wave = 16x16 tile x NITER along N. One-hot B fragments and target data
//   are built ONCE per wave; all hot-loop pointers advance by constant
//   strides so stores/loads use immediate instruction offsets.
// ---------------------------------------------------------------------------
__global__ void __launch_bounds__(256)
cost_kernel(const __hip_bfloat16* __restrict__ negprob,
            const float* __restrict__ pred_cxcywh,
            const float* __restrict__ tgt_cxcywh,
            const int*   __restrict__ labels,
            const float* __restrict__ pxyxy,
            const float* __restrict__ txyxy,
            float* __restrict__ out) {
    // per-wave staging of 16 pred rows: cx,cy,w,h | x0,y0,x1,y1  (8 floats)
    __shared__ float lds[8][16][8];

    const int tid   = threadIdx.x;
    const int wv    = tid >> 5;
    const int lane  = tid & 31;
    const int waveN = wv & 3;          // 4 tile-bands along N
    const int waveM = wv >> 2;         // 2 tiles along M
    const int half  = lane >> 4;       // 0: lanes 0-15, 1: lanes 16-31
    const int l15   = lane & 15;

    // ---- per-wave invariants: target column data + one-hot B fragments -----
    const int m   = blockIdx.y * 32 + waveM * 16 + l15;
    const int lab = labels[m];

    const f32x4 tc = *(const f32x4*)(tgt_cxcywh + (size_t)m * 4);
    const f32x4 tx = *(const f32x4*)(txyxy      + (size_t)m * 4);
    const float ta = tc.z * tc.w;

    // B-fragment (32x16 bf16): lane col = l15; element i = K (kc + half*16 + i).
    // Build dword-wise: dword j holds elements {2j, 2j+1}.
    v16bf Bfrag[3];
    #pragma unroll
    for (int c = 0; c < 3; ++c) {
        union { v16bf v; unsigned int d[8]; } Bu;
        const int rel  = lab - c * 32 - half * 16;       // hit if rel in [0,16)
        const unsigned pat  = (rel & 1) ? 0x3F800000u : 0x00003F80u;
        const int      relj = rel >> 1;                   // arithmetic: neg -> no hit
        #pragma unroll
        for (int j = 0; j < 8; ++j)
            Bu.d[j] = (relj == j) ? pat : 0u;
        Bfrag[c] = Bu.v;
    }

    // ---- hot-loop pointers: computed once, advanced by constant strides ----
    const int rowBase = blockIdx.x * ROWS_PER_BLOCK + waveN * 16;

    // A-fragment addressing (contiguous K, matches B fragment):
    //   lane half 0: elements 0..15 = K kc+0..15
    //   lane half 1: elements 0..15 = K kc+16..31
    const __hip_bfloat16* aBase =
        negprob + (size_t)(rowBase + l15) * KPAD + half * 16;

    // output base: 8 stores per tile = obase[r*MT], r*MT*4 fits 24-bit IOFFSET
    float* obase = out + (size_t)(rowBase + 8 * half) * MT + m;

    // LDS staging sources (only lanes 0..15 dereference)
    const f32x4* stageC = (const f32x4*)pred_cxcywh + (rowBase + lane);
    const f32x4* stageX = (const f32x4*)pxyxy       + (rowBase + lane);

    #pragma unroll 1
    for (int it = 0; it < NITER; ++it) {
        // ---- stage this tile's 16 pred rows into LDS (2 x b128 per row) ----
        if (lane < 16) {
            f32x4 c = *stageC;
            f32x4 x = *stageX;
            *(f32x4*)&lds[wv][lane][0] = c;
            *(f32x4*)&lds[wv][lane][4] = x;
        }

        // keep the bf16 prob stream ahead of the WMMA loop (speculative;
        // past-the-end lands inside the same workspace allocation)
        __builtin_prefetch((const void*)(aBase + 64 * KPAD), 0, 1);

        // ---- class cost: (-prob) x onehot via 3 WMMAs -----------------------
        v8f acc = {};
        #pragma unroll
        for (int c = 0; c < 3; ++c) {
            union { v16bf v; f32x4 f[2]; } A;
            A.f[0] = *(const f32x4*)(aBase + c * 32);
            A.f[1] = *(const f32x4*)(aBase + c * 32 + 8);
            acc = __builtin_amdgcn_wmma_f32_16x16x32_bf16(
                false, A.v, false, Bfrag[c], (short)0, acc, false, false);
        }

        // ---- L1 + GIoU per pair, in the WMMA C/D register layout ------------
        #pragma unroll
        for (int r = 0; r < 8; ++r) {
            const int nrow = r + 8 * half;               // row index inside tile
            const f32x4 pc = *(const f32x4*)&lds[wv][nrow][0];  // cxcywh (broadcast)
            const f32x4 px = *(const f32x4*)&lds[wv][nrow][4];  // xyxy

            // L1 cdist on cxcywh
            const float l1 = fabsf(pc.x - tc.x) + fabsf(pc.y - tc.y) +
                             fabsf(pc.z - tc.z) + fabsf(pc.w - tc.w);

            // intersection
            const float ix0 = fmaxf(px.x, tx.x), iy0 = fmaxf(px.y, tx.y);
            const float ix1 = fminf(px.z, tx.z), iy1 = fminf(px.w, tx.w);
            const float iw  = fmaxf(ix1 - ix0, 0.0f);
            const float ih  = fmaxf(iy1 - iy0, 0.0f);
            const float inter = iw * ih;
            const float uni   = pc.z * pc.w + ta - inter;   // pa + ta - inter

            // smallest enclosing box (w/h >= 0 by construction: no clamp)
            const float ex0 = fminf(px.x, tx.x), ey0 = fminf(px.y, tx.y);
            const float ex1 = fmaxf(px.z, tx.z), ey1 = fmaxf(px.w, tx.w);
            const float ea  = (ex1 - ex0) * (ey1 - ey0);

            const float giou = inter * __builtin_amdgcn_rcpf(uni)
                             - (ea - uni) * __builtin_amdgcn_rcpf(ea);

            const float val = l1 + acc[r] - giou;   // acc already holds -p[n,lab]

            obase[r * MT] = val;                     // immediate-offset store
        }

        // constant-stride pointer advance (single 64-bit add each)
        aBase  += 64 * KPAD;
        obase  += (size_t)64 * MT;
        stageC += 64;
        stageX += 64;
    }
}

// ---------------------------------------------------------------------------
// Host launcher
// ---------------------------------------------------------------------------
extern "C" void kernel_launch(void* const* d_in, const int* in_sizes, int n_in,
                              void* d_out, int out_size, void* d_ws, size_t ws_size,
                              hipStream_t stream) {
    const float* pred_logits = (const float*)d_in[0];   // [16,600,92]
    const float* pred_boxes  = (const float*)d_in[1];   // [16,600,4]
    const float* tgt_boxes   = (const float*)d_in[2];   // [2400,4]
    const int*   tgt_labels  = (const int*)d_in[3];     // [2400]
    float*       out         = (float*)d_out;           // [16,600,2400]

    // workspace carve-up (256B aligned)
    char* ws = (char*)d_ws;
    size_t off = 0;
    auto alloc = [&](size_t bytes) {
        void* p = ws + off;
        off += (bytes + 255) & ~(size_t)255;
        return p;
    };
    __hip_bfloat16* negprob = (__hip_bfloat16*)alloc((size_t)BQ * KPAD * 2); // 1.84 MB
    float* pxyxy = (float*)alloc((size_t)BQ * 4 * 4);
    float* txyxy = (float*)alloc((size_t)MT * 4 * 4);

    // 1) softmax -> -prob (bf16, K padded to 96)
    softmax_negbf16_kernel<<<BQ / 8, 256, 0, stream>>>(pred_logits, negprob);

    // 2) box prep (xyxy)
    box_prep_kernel<<<(BQ + 255) / 256, 256, 0, stream>>>(
        pred_boxes, tgt_boxes, pxyxy, txyxy);

    // 3) fused cost matrix: grid = (9600/320, 2400/32)
    dim3 grid(BQ / ROWS_PER_BLOCK, MT / 32);
    cost_kernel<<<grid, 256, 0, stream>>>(
        negprob, pred_boxes, tgt_boxes, tgt_labels,
        pxyxy, txyxy, out);
}
